// MoE_60421599920825
// MI455X (gfx1250) — matrix operand order repaired and checked by reading, not compiled, not run on hardware
//
#include <hip/hip_runtime.h>
#include <hip/hip_bf16.h>
#include <math.h>

// ---------------------------------------------------------------------------
// MoE (DeepSeek-style sigmoid gate, top-2 of 8 routed experts + shared expert)
// for MI455X / gfx1250.  Matmuls via v_wmma_f32_16x16x32_bf16 (wave32);
// global->LDS tile movement via global_load_async_to_lds_b128 (ASYNCcnt),
// issued as inline asm (portable across toolchains per CDNA5_HIP.md).
// ---------------------------------------------------------------------------

typedef __bf16 bf16;
typedef __attribute__((ext_vector_type(16))) __bf16 v16bf;
typedef __attribute__((ext_vector_type(8)))  float  v8f;

static constexpr int T    = 8192;   // B*S tokens
static constexpr int DIM  = 2048;
static constexpr int HID  = 1792;
static constexpr int NEXP = 8;
static constexpr float SCALE = 1.0f;

// GEMM tiling: block = 256 threads = 8 waves in a 4(M) x 2(N) grid.
// Block tile 128x64, K-step 32 (one WMMA K).  Wave tile 32x32 = 2x2 WMMA tiles.
static constexpr int BM = 128, BN = 64, BK = 32, PAD = 8; // 80B row stride -> bank-conflict-free

// ---------------------------------------------------------------------------
// Async global -> LDS copy (CDNA5 path), via inline asm.
// vdst = per-lane LDS byte address (low 32 bits of the flat LDS pointer,
// per ISA 10.2 aperture rules), vaddr = 64-bit global address.
// ---------------------------------------------------------------------------
__device__ __forceinline__ void g2l_b128(void* lds_dst, const void* g_src) {
#if defined(__gfx1250__)
    unsigned int       l = (unsigned int)(unsigned long long)lds_dst;
    unsigned long long g = (unsigned long long)g_src;
    asm volatile("global_load_async_to_lds_b128 %0, %1, off"
                 :: "v"(l), "v"(g) : "memory");
#else
    *(uint4*)lds_dst = *(const uint4*)g_src;   // host-pass placeholder
#endif
}
__device__ __forceinline__ void g2l_wait() {
#if defined(__gfx1250__)
    asm volatile("s_wait_asynccnt 0x0" ::: "memory");
#endif
}

// ---------------------------------------------------------------------------
// Counters init
// ---------------------------------------------------------------------------
__global__ __launch_bounds__(64) void zero_counts_kernel(int* __restrict__ cnt) {
    if (threadIdx.x < NEXP) cnt[threadIdx.x] = 0;
}

// ---------------------------------------------------------------------------
// Gate: one wave per token.  lane = (expert e = lane&7, part p = lane>>3).
// Each lane dots DIM/4 contiguous elements, reduce over p via shfl_xor,
// sigmoid, top-2 by biased score, normalized unbiased weights, atomic scatter
// into per-expert compacted token lists.
// ---------------------------------------------------------------------------
__global__ __launch_bounds__(256)
void gate_kernel(const float* __restrict__ x, const float* __restrict__ emb,
                 const float* __restrict__ bias,
                 int* __restrict__ cnt, int* __restrict__ list,
                 float* __restrict__ cw) {
    const int wid  = threadIdx.x >> 5;
    const int lane = threadIdx.x & 31;
    const int t = blockIdx.x * 8 + wid;
    if (t >= T) return;                       // wave-uniform
    const int e = lane & 7, p = lane >> 3;    // expert / quarter
    const float* xr = x   + (size_t)t * DIM + p * (DIM / 4);
    const float* er = emb + (size_t)e * DIM + p * (DIM / 4);
    float dot = 0.f;
    #pragma unroll 4
    for (int i = 0; i < DIM / 4; i += 4) {
        float4 a = *(const float4*)(xr + i);
        float4 b = *(const float4*)(er + i);
        dot += a.x * b.x + a.y * b.y + a.z * b.z + a.w * b.w;
    }
    dot += __shfl_xor(dot, 8, 32);
    dot += __shfl_xor(dot, 16, 32);
    const float sc = 1.f / (1.f + expf(-dot));    // sigmoid score for expert (lane&7)
    float s[NEXP];
    #pragma unroll
    for (int k = 0; k < NEXP; ++k) s[k] = __shfl(sc, k, 32);
    if (lane == 0) {
        float bb[NEXP];
        #pragma unroll
        for (int k = 0; k < NEXP; ++k) bb[k] = s[k] + bias[k];
        int i0 = 0; float b0 = bb[0];
        #pragma unroll
        for (int k = 1; k < NEXP; ++k) if (bb[k] > b0) { b0 = bb[k]; i0 = k; }
        int i1 = -1; float b1 = -3.0e38f;
        #pragma unroll
        for (int k = 0; k < NEXP; ++k) if (k != i0 && bb[k] > b1) { b1 = bb[k]; i1 = k; }
        const float w0 = s[i0], w1 = s[i1];
        const float inv = SCALE / (w0 + w1);
        int slot0 = atomicAdd(&cnt[i0], 1);
        list[i0 * T + slot0] = t;  cw[i0 * T + slot0] = w0 * inv;
        int slot1 = atomicAdd(&cnt[i1], 1);
        list[i1 * T + slot1] = t;  cw[i1 * T + slot1] = w1 * inv;
    }
}

// ---------------------------------------------------------------------------
// fp32 -> bf16 conversion (grid-stride, n is a multiple of 4)
// ---------------------------------------------------------------------------
__global__ __launch_bounds__(256)
void cvt_bf16_kernel(const float* __restrict__ src, bf16* __restrict__ dst, long n) {
    long i = (long)blockIdx.x * 1024 + (long)threadIdx.x * 4;
    const long stride = (long)gridDim.x * 1024;
    for (; i + 3 < n; i += stride) {
        float4 v = *(const float4*)(src + i);
        dst[i + 0] = (bf16)v.x;  dst[i + 1] = (bf16)v.y;
        dst[i + 2] = (bf16)v.z;  dst[i + 3] = (bf16)v.w;
    }
}

// ---------------------------------------------------------------------------
// WMMA fragment loads from LDS, following ISA 7.12.2 layouts (wave32).
// A (16x32 bf16): lane L -> row = L&15; elems e<8: k = (L>>4)*8 + e,
//                 elems e>=8: k = 16 + (L>>4)*8 + (e-8).
// B (32x16 bf16) stored as [N][K] in LDS: lane L -> n = L&15,
//                 same k order as A.
// ---------------------------------------------------------------------------
__device__ __forceinline__ v16bf lds_a_frag(const bf16* base, int row, int lane) {
    const bf16* p = base + (size_t)row * (BK + PAD) + ((lane >> 4) << 3);
    union { v16bf v; uint4 u[2]; } r;
    r.u[0] = *(const uint4*)p;
    r.u[1] = *(const uint4*)(p + 16);
    return r.v;
}
__device__ __forceinline__ v16bf lds_b_frag(const bf16* base, int row, int lane) {
    const bf16* p = base + (size_t)row * (BK + PAD) + ((lane >> 4) << 3);
    union { v16bf v; uint4 u[2]; } r;
    r.u[0] = *(const uint4*)p;
    r.u[1] = *(const uint4*)(p + 16);
    return r.v;
}

// ---------------------------------------------------------------------------
// FFN stage 1: h = silu(Xe @ Wg^T) * (Xe @ Wd^T), bf16 out, compacted rows.
// Dual accumulators share the A fragment.  listB==nullptr => shared expert
// (identity rows, count = T).  blockIdx.z = expert.
// ---------------------------------------------------------------------------
__global__ __launch_bounds__(256, 1)
void ffn1_kernel(const bf16* __restrict__ xbf,
                 const bf16* __restrict__ WgB, const bf16* __restrict__ WdB,
                 bf16* __restrict__ HoutB,
                 const int* __restrict__ cnt, const int* __restrict__ listB) {
    __shared__ bf16 As[2][BM][BK + PAD];
    __shared__ bf16 Bg[2][BN][BK + PAD];
    __shared__ bf16 Bd[2][BN][BK + PAD];

    const int e = blockIdx.z;
    const bf16* wg = WgB + (size_t)e * HID * DIM;
    const bf16* wd = WdB + (size_t)e * HID * DIM;
    bf16* hout = HoutB + (size_t)e * T * HID;
    const int* lst = listB ? (listB + (size_t)e * T) : nullptr;
    const int count = cnt ? cnt[e] : T;

    const int m0 = blockIdx.y * BM;
    if (m0 >= count) return;                     // block-uniform
    const int n0 = blockIdx.x * BN;

    const int tid  = threadIdx.x;
    const int lane = tid & 31;
    const int wid  = tid >> 5;
    const int wm   = wid & 3;                    // wave row group (x32)
    const int wn   = wid >> 2;                   // wave col group (x32)

    // Per-thread global->LDS chunk assignment (16B chunks)
    int arow[2], acol[2], atok[2];
    #pragma unroll
    for (int i = 0; i < 2; ++i) {
        const int chunk = tid + 256 * i;         // 512 A chunks
        arow[i] = chunk >> 2;
        acol[i] = (chunk & 3) << 3;
        const int gm = m0 + arow[i];
        atok[i] = (gm < count) ? (lst ? lst[gm] : gm) : 0;
    }
    const int brow = tid >> 2;                   // 256 B chunks (per matrix)
    const int bcol = (tid & 3) << 3;
    const bf16* wgp = wg + (size_t)(n0 + brow) * DIM + bcol;
    const bf16* wdp = wd + (size_t)(n0 + brow) * DIM + bcol;

    constexpr int KT = DIM / BK;                 // 64 K-steps

    auto issue_tile = [&](int buf, int k0) {
        #pragma unroll
        for (int i = 0; i < 2; ++i)
            g2l_b128(&As[buf][arow[i]][acol[i]],
                     xbf + (size_t)atok[i] * DIM + k0 + acol[i]);
        g2l_b128(&Bg[buf][brow][bcol], wgp + k0);
        g2l_b128(&Bd[buf][brow][bcol], wdp + k0);
    };

    issue_tile(0, 0);
    g2l_wait();
    __syncthreads();

    v8f accg[2][2], accd[2][2];
    #pragma unroll
    for (int i = 0; i < 2; ++i)
        #pragma unroll
        for (int j = 0; j < 2; ++j) { accg[i][j] = (v8f){}; accd[i][j] = (v8f){}; }

    for (int kt = 0; kt < KT; ++kt) {
        const int cur = kt & 1, nxt = cur ^ 1;
        if (kt + 1 < KT) {
            issue_tile(nxt, (kt + 1) * BK);      // async copy overlaps the WMMAs below
            if (kt + 2 < KT) {                   // global_prefetch_b8 of the tile after next
                __builtin_prefetch(wgp + (kt + 2) * BK, 0, 1);
                __builtin_prefetch(wdp + (kt + 2) * BK, 0, 1);
            }
        }
        v16bf af[2], bgf[2], bdf[2];
        #pragma unroll
        for (int i = 0; i < 2; ++i)
            af[i] = lds_a_frag(&As[cur][0][0], wm * 32 + i * 16 + (lane & 15), lane);
        #pragma unroll
        for (int j = 0; j < 2; ++j) {
            bgf[j] = lds_b_frag(&Bg[cur][0][0], wn * 32 + j * 16 + (lane & 15), lane);
            bdf[j] = lds_b_frag(&Bd[cur][0][0], wn * 32 + j * 16 + (lane & 15), lane);
        }
        #pragma unroll
        for (int i = 0; i < 2; ++i)
            #pragma unroll
            for (int j = 0; j < 2; ++j) {
                accg[i][j] = __builtin_amdgcn_wmma_f32_16x16x32_bf16(
                    false, af[i], false, bgf[j], (short)0, accg[i][j], false, false);
                accd[i][j] = __builtin_amdgcn_wmma_f32_16x16x32_bf16(
                    false, af[i], false, bdf[j], (short)0, accd[i][j], false, false);
            }
        if (kt + 1 < KT) {
            g2l_wait();                           // drain this wave's async copies
            __syncthreads();                      // writes visible to all waves
        }
    }

    // Epilogue: h = silu(g) * d  (C/D layout: elem r -> M = r + 8*(lane>=16))
    #pragma unroll
    for (int i = 0; i < 2; ++i)
        #pragma unroll
        for (int j = 0; j < 2; ++j)
            #pragma unroll
            for (int r = 0; r < 8; ++r) {
                const int m = m0 + wm * 32 + i * 16 + r + ((lane >> 4) << 3);
                if (m < count) {
                    const int col = n0 + wn * 32 + j * 16 + (lane & 15);
                    const float g = accg[i][j][r], d = accd[i][j][r];
                    const float h = (g / (1.f + expf(-g))) * d;
                    hout[(size_t)m * HID + col] = (bf16)h;
                }
            }
}

// ---------------------------------------------------------------------------
// FFN stage 2: out[token] (+)= cw * (h @ Wu^T).
// listB==nullptr => shared expert: plain store (initializes d_out).
// Routed experts: unsafeAtomicAdd (hardware global_atomic_add_f32).
// ---------------------------------------------------------------------------
__global__ __launch_bounds__(256, 1)
void ffn2_kernel(const bf16* __restrict__ HB, const bf16* __restrict__ WuB,
                 float* __restrict__ out,
                 const int* __restrict__ cnt, const int* __restrict__ listB,
                 const float* __restrict__ cwB) {
    __shared__ bf16 As[2][BM][BK + PAD];
    __shared__ bf16 Bu[2][BN][BK + PAD];

    const int e = blockIdx.z;
    const bf16* h  = HB  + (size_t)e * T * HID;
    const bf16* wu = WuB + (size_t)e * DIM * HID;
    const int*   lst = listB ? (listB + (size_t)e * T) : nullptr;
    const float* cw  = cwB   ? (cwB   + (size_t)e * T) : nullptr;
    const int count = cnt ? cnt[e] : T;

    const int m0 = blockIdx.y * BM;
    if (m0 >= count) return;
    const int n0 = blockIdx.x * BN;

    const int tid  = threadIdx.x;
    const int lane = tid & 31;
    const int wid  = tid >> 5;
    const int wm   = wid & 3;
    const int wn   = wid >> 2;

    int arow[2], acol[2];
    #pragma unroll
    for (int i = 0; i < 2; ++i) {
        const int chunk = tid + 256 * i;
        arow[i] = chunk >> 2;
        acol[i] = (chunk & 3) << 3;
    }
    const int brow = tid >> 2;
    const int bcol = (tid & 3) << 3;
    const bf16* wup = wu + (size_t)(n0 + brow) * HID + bcol;

    constexpr int KT = HID / BK;                 // 56 K-steps

    auto issue_tile = [&](int buf, int k0) {
        #pragma unroll
        for (int i = 0; i < 2; ++i)
            g2l_b128(&As[buf][arow[i]][acol[i]],
                     h + (size_t)(m0 + arow[i]) * HID + k0 + acol[i]);
        g2l_b128(&Bu[buf][brow][bcol], wup + k0);
    };

    issue_tile(0, 0);
    g2l_wait();
    __syncthreads();

    v8f acc[2][2];
    #pragma unroll
    for (int i = 0; i < 2; ++i)
        #pragma unroll
        for (int j = 0; j < 2; ++j) acc[i][j] = (v8f){};

    for (int kt = 0; kt < KT; ++kt) {
        const int cur = kt & 1, nxt = cur ^ 1;
        if (kt + 1 < KT) {
            issue_tile(nxt, (kt + 1) * BK);
            if (kt + 2 < KT)
                __builtin_prefetch(wup + (kt + 2) * BK, 0, 1);
        }
        v16bf af[2], bf[2];
        #pragma unroll
        for (int i = 0; i < 2; ++i)
            af[i] = lds_a_frag(&As[cur][0][0], wm * 32 + i * 16 + (lane & 15), lane);
        #pragma unroll
        for (int j = 0; j < 2; ++j)
            bf[j] = lds_b_frag(&Bu[cur][0][0], wn * 32 + j * 16 + (lane & 15), lane);
        #pragma unroll
        for (int i = 0; i < 2; ++i)
            #pragma unroll
            for (int j = 0; j < 2; ++j)
                acc[i][j] = __builtin_amdgcn_wmma_f32_16x16x32_bf16(
                    false, af[i], false, bf[j], (short)0, acc[i][j], false, false);
        if (kt + 1 < KT) {
            g2l_wait();
            __syncthreads();
        }
    }

    #pragma unroll
    for (int i = 0; i < 2; ++i)
        #pragma unroll
        for (int j = 0; j < 2; ++j)
            #pragma unroll
            for (int r = 0; r < 8; ++r) {
                const int m = m0 + wm * 32 + i * 16 + r + ((lane >> 4) << 3);
                if (m < count) {
                    const int col = n0 + wn * 32 + j * 16 + (lane & 15);
                    const float v = acc[i][j][r];
                    if (lst) {
                        unsafeAtomicAdd(&out[(size_t)lst[m] * DIM + col], cw[m] * v);
                    } else {
                        out[(size_t)m * DIM + col] = v;
                    }
                }
            }
}

// ---------------------------------------------------------------------------
// Host launcher
// ---------------------------------------------------------------------------
extern "C" void kernel_launch(void* const* d_in, const int* in_sizes, int n_in,
                              void* d_out, int out_size, void* d_ws, size_t ws_size,
                              hipStream_t stream) {
    (void)in_sizes; (void)n_in; (void)out_size; (void)ws_size;
    const float* x    = (const float*)d_in[0];
    const float* emb  = (const float*)d_in[1];
    const float* bias = (const float*)d_in[2];
    const float* Wg   = (const float*)d_in[3];
    const float* Wd   = (const float*)d_in[4];
    const float* Wu   = (const float*)d_in[5];
    const float* sWg  = (const float*)d_in[6];
    const float* sWd  = (const float*)d_in[7];
    const float* sWu  = (const float*)d_in[8];
    float* out = (float*)d_out;

    char* ws = (char*)d_ws;
    size_t off = 0;
    auto carve = [&](size_t bytes) -> char* {
        char* p = ws + off;
        off = (off + bytes + 255) & ~(size_t)255;
        return p;
    };
    int*   cnt  = (int*)  carve((size_t)NEXP * sizeof(int));
    int*   list = (int*)  carve((size_t)NEXP * T * sizeof(int));
    float* cw   = (float*)carve((size_t)NEXP * T * sizeof(float));
    bf16*  xbf  = (bf16*) carve((size_t)T * DIM * sizeof(bf16));
    bf16*  wgb  = (bf16*) carve((size_t)NEXP * HID * DIM * sizeof(bf16));
    bf16*  wdb  = (bf16*) carve((size_t)NEXP * HID * DIM * sizeof(bf16));
    bf16*  wub  = (bf16*) carve((size_t)NEXP * DIM * HID * sizeof(bf16));
    bf16*  swgb = (bf16*) carve((size_t)HID * DIM * sizeof(bf16));
    bf16*  swdb = (bf16*) carve((size_t)HID * DIM * sizeof(bf16));
    bf16*  swub = (bf16*) carve((size_t)DIM * HID * sizeof(bf16));
    bf16*  hr   = (bf16*) carve((size_t)NEXP * T * HID * sizeof(bf16));
    bf16*  hs   = (bf16*) carve((size_t)T * HID * sizeof(bf16));

    zero_counts_kernel<<<1, 64, 0, stream>>>(cnt);
    gate_kernel<<<T / 8, 256, 0, stream>>>(x, emb, bias, cnt, list, cw);

    cvt_bf16_kernel<<<2048, 256, 0, stream>>>(x,   xbf,  (long)T * DIM);
    cvt_bf16_kernel<<<2048, 256, 0, stream>>>(Wg,  wgb,  (long)NEXP * HID * DIM);
    cvt_bf16_kernel<<<2048, 256, 0, stream>>>(Wd,  wdb,  (long)NEXP * HID * DIM);
    cvt_bf16_kernel<<<2048, 256, 0, stream>>>(Wu,  wub,  (long)NEXP * DIM * HID);
    cvt_bf16_kernel<<<512,  256, 0, stream>>>(sWg, swgb, (long)HID * DIM);
    cvt_bf16_kernel<<<512,  256, 0, stream>>>(sWd, swdb, (long)HID * DIM);
    cvt_bf16_kernel<<<512,  256, 0, stream>>>(sWu, swub, (long)DIM * HID);

    // Routed experts stage 1 (compacted token lists), then shared expert stage 1
    dim3 g1(HID / BN, T / BM, NEXP);     // 28 x 64 x 8
    ffn1_kernel<<<g1, 256, 0, stream>>>(xbf, wgb, wdb, hr, cnt, list);
    dim3 g1s(HID / BN, T / BM, 1);
    ffn1_kernel<<<g1s, 256, 0, stream>>>(xbf, swgb, swdb, hs, nullptr, nullptr);

    // Stage 2: shared expert first (plain stores initialize d_out),
    // then routed experts accumulate atomically (stream-ordered).
    dim3 g2s(DIM / BN, T / BM, 1);       // 32 x 64
    ffn2_kernel<<<g2s, 256, 0, stream>>>(hs, swub, out, nullptr, nullptr, nullptr);
    dim3 g2(DIM / BN, T / BM, NEXP);     // 32 x 64 x 8
    ffn2_kernel<<<g2, 256, 0, stream>>>(hr, wub, out, cnt, list, cw);
}
